// NodeRegressor_46943992545635
// MI455X (gfx1250) — compile-verified, matrix operand order checked by reading, not run on hardware
//
#include <hip/hip_runtime.h>

// Shapes (fixed by reference): D=256, H=8, DH=32, M=8, B=1
#define DFEAT 256
#define HEADS 8
#define DHEAD 32
#define MARKS 8

typedef __attribute__((ext_vector_type(16))) _Float16 v16h;
typedef __attribute__((ext_vector_type(8)))  _Float16 v8h;
typedef __attribute__((ext_vector_type(8)))  float    v8f;

// ---- order-preserving float<->uint encoding for atomic max ----
__device__ __forceinline__ unsigned encf(float f) {
    unsigned u = __float_as_uint(f);
    return (u & 0x80000000u) ? ~u : (u | 0x80000000u);
}
__device__ __forceinline__ float decf(unsigned e) {
    return (e & 0x80000000u) ? __uint_as_float(e & 0x7FFFFFFFu)
                             : __uint_as_float(~e);
}

// ---------------------------------------------------------------------------
// init: d_out = h_dst (fused residual), zero g/den/mx, fp32 -> f16 activations
// ---------------------------------------------------------------------------
__global__ void k_init(const float* __restrict__ h_src,
                       const float* __restrict__ h_dst,
                       float* __restrict__ out, float* __restrict__ g,
                       float* __restrict__ den, unsigned* __restrict__ mx,
                       _Float16* __restrict__ hsrc_h,
                       _Float16* __restrict__ hdst_h, int NF, int NH) {
    int i = blockIdx.x * blockDim.x + threadIdx.x;
    if (i < NF) {
        float hd = h_dst[i];
        out[i] = hd;          // residual base
        g[i]   = 0.f;
        hdst_h[i] = (_Float16)hd;
        hsrc_h[i] = (_Float16)h_src[i];
    }
    if (i < NH) { den[i] = 0.f; mx[i] = 0u; }
}

// weights fp32 -> f16, packed [Wq|Wk|Wt|Wv] to match GEMM z-index
__global__ void k_convw(const float* __restrict__ Wq, const float* __restrict__ Wk,
                        const float* __restrict__ Wt, const float* __restrict__ Wv,
                        _Float16* __restrict__ Wh) {
    int i = blockIdx.x * blockDim.x + threadIdx.x;   // 0 .. 4*65536-1
    int which = i >> 16, loc = i & 65535;
    const float* s = (which == 0) ? Wq : (which == 1) ? Wk : (which == 2) ? Wt : Wv;
    Wh[i] = (_Float16)s[loc];
}

// ---------------------------------------------------------------------------
// WMMA GEMM:  C[z] = A[z] @ W[z]^T
//   z=0: Q = h_dst@Wq^T ; z=1: K = h_src@Wk^T ; z=2: T ; z=3: V
// wave computes one 16x16 tile; K-loop of 8 x wmma_f32_16x16x32_f16
// ---------------------------------------------------------------------------
__global__ void __launch_bounds__(256) k_gemm(
        const _Float16* __restrict__ hsrc_h,
        const _Float16* __restrict__ hdst_h,
        const _Float16* __restrict__ Wh,
        float* __restrict__ outQKTV, int N) {
    const int z = blockIdx.z;
    const _Float16* A = (z == 0) ? hdst_h : hsrc_h;
    const _Float16* W = Wh + (size_t)z * (DFEAT * DFEAT);
    float* C = outQKTV + (size_t)z * (size_t)N * DFEAT;

    const int lane = threadIdx.x & 31;
    const int wave = threadIdx.x >> 5;
    const int half = lane >> 4;        // 0: K{0..7,16..23}, 1: K{8..15,24..31}
    const int r16  = lane & 15;
    const int rowBase = blockIdx.x * 16;
    const int colBase = (blockIdx.y * 8 + wave) * 16;

    int arow = rowBase + r16; if (arow >= N) arow = N - 1;   // clamp loads
    const _Float16* Ap = A + (size_t)arow * DFEAT;
    const _Float16* Bp = W + (size_t)(colBase + r16) * DFEAT; // B col n = W row n

    v8f acc = {};
#pragma unroll
    for (int k = 0; k < DFEAT; k += 32) {
        const int k0 = k + half * 8;
        v8h alo = *(const v8h*)(Ap + k0);
        v8h ahi = *(const v8h*)(Ap + k0 + 16);
        v8h blo = *(const v8h*)(Bp + k0);
        v8h bhi = *(const v8h*)(Bp + k0 + 16);
        v16h a, b;
#pragma unroll
        for (int i = 0; i < 8; ++i) {
            a[i] = alo[i]; a[i + 8] = ahi[i];
            b[i] = blo[i]; b[i + 8] = bhi[i];
        }
        // (neg_a, A, neg_b, B, c_mod, C, reuse_a, reuse_b)
        acc = __builtin_amdgcn_wmma_f32_16x16x32_f16(
            false, a, false, b, (short)0, acc, false, false);
    }
    const int col = colBase + r16;
#pragma unroll
    for (int i = 0; i < 8; ++i) {
        const int row = rowBase + i + half * 8;   // VGPR i -> M=i (lanes<16) / i+8
        if (row < N) C[(size_t)row * DFEAT + col] = acc[i];
    }
}

// WeQ[n,h] = sum_dh We[h*32+dh] * Q[n,h,dh]   (rank-1 ef-feature folded into logit)
__global__ void k_weq(const float* __restrict__ Qf, const float* __restrict__ We,
                      float* __restrict__ WeQ, int NH) {
    int t = blockIdx.x * blockDim.x + threadIdx.x;
    if (t >= NH) return;
    int n = t >> 3, h = t & 7;
    const float* q = Qf + (size_t)n * DFEAT + h * DHEAD;
    const float* w = We + h * DHEAD;
    float s = 0.f;
#pragma unroll
    for (int d = 0; d < DHEAD; ++d) s += q[d] * w[d];
    WeQ[t] = s;
}

// logits + segment max (ordered-uint atomicMax)
__global__ void k_logit(const float* __restrict__ Qf, const float* __restrict__ Kf,
                        const float* __restrict__ ef, const float* __restrict__ WeQ,
                        const int* __restrict__ src, const int* __restrict__ dst,
                        float* __restrict__ ebuf, unsigned* __restrict__ mx, int EH) {
    int t = blockIdx.x * blockDim.x + threadIdx.x;
    if (t >= EH) return;
    int e = t >> 3, h = t & 7;
    int s = src[e], d = dst[e];
    const float4* kq = (const float4*)(Kf + (size_t)s * DFEAT + h * DHEAD);
    const float4* qq = (const float4*)(Qf + (size_t)d * DFEAT + h * DHEAD);
    float dot = 0.f;
#pragma unroll
    for (int i = 0; i < 8; ++i) {
        float4 a = kq[i], b = qq[i];
        dot += a.x * b.x + a.y * b.y + a.z * b.z + a.w * b.w;
    }
    float logit = (dot + ef[e] * WeQ[d * 8 + h]) * 0.17677669529663687f; // 1/sqrt(32)
    ebuf[t] = logit;
    atomicMax(&mx[d * 8 + h], encf(logit));
}

// ex = exp(e - mx[dst]); den += ex
__global__ void k_exp(const int* __restrict__ dst, const unsigned* __restrict__ mx,
                      float* __restrict__ ebuf, float* __restrict__ den, int EH) {
    int t = blockIdx.x * blockDim.x + threadIdx.x;
    if (t >= EH) return;
    int e = t >> 3, h = t & 7, d = dst[e];
    float ex = __expf(ebuf[t] - decf(mx[d * 8 + h]));
    ebuf[t] = ex;
    atomicAdd(&den[d * 8 + h], ex);
}

// ea = ex/den[dst]; g[dst] += T[src]*ea
__global__ void k_gagg(const float* __restrict__ Tf, const float* __restrict__ den,
                       const int* __restrict__ src, const int* __restrict__ dst,
                       float* __restrict__ ebuf, float* __restrict__ g, int EH) {
    int t = blockIdx.x * blockDim.x + threadIdx.x;
    if (t >= EH) return;
    int e = t >> 3, h = t & 7;
    int s = src[e], d = dst[e];
    float ea = ebuf[t] / den[d * 8 + h];
    ebuf[t] = ea;                                   // keep for V-aggregation
    const float* tv = Tf + (size_t)s * DFEAT + h * DHEAD;
    float* gp = g + (size_t)d * DFEAT + h * DHEAD;
#pragma unroll
    for (int i = 0; i < DHEAD; ++i) atomicAdd(&gp[i], tv[i] * ea);
}

// ---------------------------------------------------------------------------
// Hawkes mark intensities: block per node, wave per head, Wi resident in LDS.
// stride-33 LDS rows -> bank-conflict-free on 64 banks.
// ---------------------------------------------------------------------------
__global__ void __launch_bounds__(256) k_hawkes(
        const float* __restrict__ g, const float* __restrict__ ts,
        const float* __restrict__ Wi, const float* __restrict__ bi,
        const float* __restrict__ wgt, const float* __restrict__ scl,
        float* __restrict__ fi, int N) {
    __shared__ float sWi[DFEAT * 33];
    __shared__ float sbi[DFEAT];
    __shared__ float swt[MARKS * DHEAD];
    __shared__ float sx[HEADS * 33];
    __shared__ float sscale[MARKS];

    const int n = blockIdx.x;
    if (n >= N) return;                 // uniform over block
    const int t = threadIdx.x;
    for (int i = t; i < DFEAT * 33; i += 256) sWi[i] = Wi[i];
    sbi[t] = bi[t];
    swt[t] = wgt[t];
    if (t < MARKS) sscale[t] = __expf(scl[t]);

    const int wave = t >> 5, lane = t & 31;         // wave == head
    sx[wave * 33 + lane] = g[(size_t)n * DFEAT + wave * DHEAD + lane];
    if (lane == 0) sx[wave * 33 + 32] = ts[n];
    __syncthreads();

    float p[MARKS];
    const float* x = &sx[wave * 33];
#pragma unroll
    for (int m = 0; m < MARKS; ++m) {
        const int r = m * DHEAD + lane;
        float a = sbi[r];
        const float* wrow = &sWi[r * 33];
#pragma unroll
        for (int j = 0; j < 33; ++j) a += wrow[j] * x[j];
        float mu = 1.f / (1.f + __expf(-a));
        p[m] = mu * swt[m * DHEAD + lane];
    }
#pragma unroll
    for (int m = 0; m < MARKS; ++m)
        for (int off = 16; off; off >>= 1)
            p[m] += __shfl_xor(p[m], off, 32);
    if (lane < MARKS) {
        const int m = lane;
        float sc  = sscale[m];
        float raw = p[m] / sc;
        float sp  = fmaxf(raw, 0.f) + log1pf(__expf(-fabsf(raw)));  // softplus
        fi[((size_t)n * HEADS + wave) * MARKS + m] = sc * sp;
    }
}

// ee = (fi[src]·marks[dst]) * ea; out[dst] += V[src]*ee  (residual already in out)
__global__ void k_vagg(const float* __restrict__ Vf, const float* __restrict__ fi,
                       const float* __restrict__ marks,
                       const int* __restrict__ src, const int* __restrict__ dst,
                       const float* __restrict__ ebuf, float* __restrict__ out, int EH) {
    int t = blockIdx.x * blockDim.x + threadIdx.x;
    if (t >= EH) return;
    int e = t >> 3, h = t & 7;
    int s = src[e], d = dst[e];
    const float* fp = fi + ((size_t)s * HEADS + h) * MARKS;
    const float* mk = marks + (size_t)d * MARKS;
    float sum = 0.f;
#pragma unroll
    for (int m = 0; m < MARKS; ++m) sum += fp[m] * mk[m];
    float ee = sum * ebuf[t];
    const float* vv = Vf + (size_t)s * DFEAT + h * DHEAD;
    float* op = out + (size_t)d * DFEAT + h * DHEAD;
#pragma unroll
    for (int i = 0; i < DHEAD; ++i) atomicAdd(&op[i], vv[i] * ee);
}

// ---------------------------------------------------------------------------
extern "C" void kernel_launch(void* const* d_in, const int* in_sizes, int n_in,
                              void* d_out, int out_size, void* d_ws, size_t ws_size,
                              hipStream_t stream) {
    const float* h_src   = (const float*)d_in[0];
    const float* h_dst   = (const float*)d_in[1];
    const float* tspan   = (const float*)d_in[2];
    const float* marks   = (const float*)d_in[3];
    const float* ef      = (const float*)d_in[4];
    const float* Wq      = (const float*)d_in[5];
    const float* Wk      = (const float*)d_in[6];
    const float* Wv      = (const float*)d_in[7];   // NOTE: Wv before Wt in dict
    const float* Wt      = (const float*)d_in[8];
    const float* We      = (const float*)d_in[9];
    const float* Wi      = (const float*)d_in[10];
    const float* bi      = (const float*)d_in[11];
    const float* wgt_i   = (const float*)d_in[12];
    const float* scl_i   = (const float*)d_in[13];
    const int*   src     = (const int*)d_in[14];
    const int*   dst     = (const int*)d_in[15];

    const int N  = in_sizes[2];          // timespans: N*B*1
    const int E  = in_sizes[14];
    const int NF = N * DFEAT;
    const int NH = N * HEADS;
    const int EH = E * HEADS;

    // workspace carve-out (256B aligned regions)
    size_t off = 0;
    char* base = (char*)d_ws;
    auto carve = [&](size_t bytes) -> void* {
        void* p = base + off;
        off += (bytes + 255) & ~(size_t)255;
        return p;
    };
    float*    QKTV   = (float*)carve((size_t)4 * NF * sizeof(float)); // Q|K|T|V
    float*    g      = (float*)carve((size_t)NF * sizeof(float));
    float*    ebuf   = (float*)carve((size_t)EH * sizeof(float));
    float*    fi     = (float*)carve((size_t)NH * MARKS * sizeof(float));
    float*    WeQ    = (float*)carve((size_t)NH * sizeof(float));
    float*    den    = (float*)carve((size_t)NH * sizeof(float));
    unsigned* mx     = (unsigned*)carve((size_t)NH * sizeof(unsigned));
    _Float16* hsrc_h = (_Float16*)carve((size_t)NF * sizeof(_Float16));
    _Float16* hdst_h = (_Float16*)carve((size_t)NF * sizeof(_Float16));
    _Float16* Wh     = (_Float16*)carve((size_t)4 * DFEAT * DFEAT * sizeof(_Float16));

    float* Qf = QKTV;
    float* Kf = QKTV + (size_t)1 * NF;
    float* Tf = QKTV + (size_t)2 * NF;
    float* Vf = QKTV + (size_t)3 * NF;

    const int TB = 256;
    k_init<<<(NF + TB - 1) / TB, TB, 0, stream>>>(h_src, h_dst, (float*)d_out, g,
                                                  den, mx, hsrc_h, hdst_h, NF, NH);
    k_convw<<<(4 * DFEAT * DFEAT) / TB, TB, 0, stream>>>(Wq, Wk, Wt, Wv, Wh);

    dim3 ggrid((N + 15) / 16, 2, 4);     // row tiles x col groups(2x8 waves) x {Q,K,T,V}
    k_gemm<<<ggrid, TB, 0, stream>>>(hsrc_h, hdst_h, Wh, QKTV, N);

    k_weq<<<(NH + TB - 1) / TB, TB, 0, stream>>>(Qf, We, WeQ, NH);
    k_logit<<<(EH + TB - 1) / TB, TB, 0, stream>>>(Qf, Kf, ef, WeQ, src, dst, ebuf, mx, EH);
    k_exp<<<(EH + TB - 1) / TB, TB, 0, stream>>>(dst, mx, ebuf, den, EH);
    k_gagg<<<(EH + TB - 1) / TB, TB, 0, stream>>>(Tf, den, src, dst, ebuf, g, EH);
    k_hawkes<<<N, TB, 0, stream>>>(g, tspan, Wi, bi, wgt_i, scl_i, fi, N);
    k_vagg<<<(EH + TB - 1) / TB, TB, 0, stream>>>(Vf, fi, marks, src, dst, ebuf,
                                                  (float*)d_out, EH);
}
